// TalaThread_70463233458678
// MI455X (gfx1250) — compile-verified, matrix operand order checked by reading, not run on hardware
//
#include <hip/hip_runtime.h>
#include <hip/hip_bf16.h>

typedef __attribute__((ext_vector_type(16))) _Float16 v16h;
typedef __attribute__((ext_vector_type(8)))  float    v8f;
typedef __attribute__((ext_vector_type(4)))  int      v4i;

#define Bsz 4
#define Tsz 4096
#define Dsz 128
#define BT  (Bsz * Tsz)        // 16384
#define NC1 768                // 3D (xp) + D (Q) + D (K) + D (V)
#define CYC 32

// ---- CDNA5 async global->LDS path (guarded; fallback = prefetch + global) --
#if defined(__has_builtin)
#if __has_builtin(__builtin_amdgcn_global_load_async_to_lds_b128)
#define ASYNC_LDS 1
#endif
#endif

#ifdef ASYNC_LDS
typedef __attribute__((address_space(1))) v4i as1_v4i;   // global
typedef __attribute__((address_space(3))) v4i as3_v4i;   // LDS
__device__ __forceinline__ void wait_async0() {
#if __has_builtin(__builtin_amdgcn_s_wait_asynccnt)
    __builtin_amdgcn_s_wait_asynccnt(0);
#else
    asm volatile("s_wait_asynccnt 0x0" ::: "memory");
#endif
}
#endif

// ---------------------------------------------------------------------------
// Prep: fp32 -> fp16 conversions & weight packing
// wcat rows: [0,384)=w_ih  [384,512)=wq  [512,640)=wk  [640,768)=wv
// ---------------------------------------------------------------------------
__global__ __launch_bounds__(256) void k_prep(
    const float* __restrict__ x, const float* __restrict__ w_ih,
    const float* __restrict__ wq, const float* __restrict__ wk,
    const float* __restrict__ wv, const float* __restrict__ b_ih,
    const float* __restrict__ wp,
    _Float16* __restrict__ xf16, _Float16* __restrict__ wcat,
    float* __restrict__ biascat, _Float16* __restrict__ wp16)
{
    int i = blockIdx.x * blockDim.x + threadIdx.x;
    if (i < BT * Dsz)   xf16[i] = (_Float16)x[i];
    if (i < 384 * Dsz)  wcat[i] = (_Float16)w_ih[i];
    if (i < Dsz * Dsz) {
        wcat[384 * Dsz + i] = (_Float16)wq[i];
        wcat[512 * Dsz + i] = (_Float16)wk[i];
        wcat[640 * Dsz + i] = (_Float16)wv[i];
    }
    if (i < NC1) biascat[i] = (i < 384) ? b_ih[i] : 0.0f;
    if (i < Dsz * 256) wp16[i] = (_Float16)wp[i];
}

// ---------------------------------------------------------------------------
// GEMM1: P[16384 x 768] = xf16[16384 x 128] @ wcat^T  (+bias on xp cols)
// 8 waves / block, one 16x16 tile per wave, K = 4 chunks of 32.
// ---------------------------------------------------------------------------
__global__ __launch_bounds__(256) void k_gemm1(
    const _Float16* __restrict__ xf16, const _Float16* __restrict__ wcat,
    const float* __restrict__ biascat, float* __restrict__ P)
{
    int wave = threadIdx.x >> 5, lane = threadIdx.x & 31;
    int tile = blockIdx.x * 8 + wave;            // 1024 * 48 tiles
    int mt = tile / 48, nt = tile % 48;
    int lrow = lane & 15, hi = lane >> 4;

    v8f acc;
#pragma unroll
    for (int i = 0; i < 8; i++) acc[i] = 0.0f;

    const _Float16* arow = xf16 + (size_t)(mt * 16 + lrow) * Dsz + hi * 16;
    const _Float16* brow = wcat + (size_t)(nt * 16 + lrow) * Dsz + hi * 16;
#pragma unroll
    for (int c = 0; c < 4; c++) {
        v16h a = *(const v16h*)(arow + c * 32);
        v16h b = *(const v16h*)(brow + c * 32);
        acc = __builtin_amdgcn_wmma_f32_16x16x32_f16(false, a, false, b,
                                                     (short)0, acc, false, false);
    }
    int col = nt * 16 + lrow;
    float bias = biascat[col];
#pragma unroll
    for (int r = 0; r < 8; r++) {
        int row = mt * 16 + r + hi * 8;
        P[(size_t)row * NC1 + col] = acc[r] + bias;
    }
}

// ---------------------------------------------------------------------------
// GRU scan: one block per batch, 24 waves (768 thr), w_hh resident as f16
// WMMA B operands in registers; h: fp32 master + f16 mirror in LDS.
// xp rows double-buffered into LDS via GLOBAL_LOAD_ASYNC_TO_LDS (ASYNCcnt).
// ---------------------------------------------------------------------------
__global__ __launch_bounds__(768) void k_gru(
    const float* __restrict__ P, const float* __restrict__ w_hh,
    const float* __restrict__ b_hh, const float* __restrict__ h0,
    _Float16* __restrict__ comb, float* __restrict__ hidden)
{
    int b = blockIdx.x;
    int tid = threadIdx.x, wave = tid >> 5, lane = tid & 31;
    int lrow = lane & 15, hi = lane >> 4;

    __shared__ __align__(32) _Float16 h16[Dsz];
    __shared__ float h32[Dsz];
    __shared__ float gh[384];
    __shared__ __align__(16) float xpbuf[2][384];

    // Preload this wave's 16 columns of w_hh as 4 packed f16 B chunks.
    v16h breg[4];
    int ncol = wave * 16 + lrow;
#pragma unroll
    for (int c = 0; c < 4; c++)
#pragma unroll
        for (int i = 0; i < 16; i++)
            breg[c][i] = (_Float16)w_hh[(size_t)ncol * Dsz + c * 32 + hi * 16 + i];

    float bhh = b_hh[wave * 16 + lrow];

    if (tid < Dsz) {
        float h = h0[b * Dsz + tid];
        h32[tid] = h;
        h16[tid] = (_Float16)h;
    }
    // stage xp row for t = 0
    if (tid < 384) xpbuf[0][tid] = P[(size_t)b * Tsz * NC1 + tid];
    __syncthreads();

    for (int t = 0; t < Tsz; t++) {
        v8f acc;
#pragma unroll
        for (int i = 0; i < 8; i++) acc[i] = 0.0f;
#pragma unroll
        for (int c = 0; c < 4; c++) {
            v16h a;
#pragma unroll
            for (int i = 0; i < 16; i++) a[i] = (_Float16)0;
            if (lrow == 0)  // only A row 0 (= h) is nonzero
                a = *(const v16h*)&h16[c * 32 + hi * 16];
            acc = __builtin_amdgcn_wmma_f32_16x16x32_f16(false, a, false, breg[c],
                                                         (short)0, acc, false, false);
        }
        if (lane < 16) gh[wave * 16 + lane] = acc[0] + bhh;  // M=0 lives in VGPR0, lanes 0-15

#ifdef ASYNC_LDS
        // issuing waves drain ASYNCcnt so the t-buffer is valid before the
        // barrier publishes it to the gating waves
        if (wave >= 4 && wave < 7) wait_async0();
#endif
        __syncthreads();  // barrier 1: gh (and staged xp) visible

#ifdef ASYNC_LDS
        if (t + 1 < Tsz && wave >= 4 && wave < 7) {
            int j = tid - 128;  // 0..95, 16 bytes each (96*16B = 1536B row)
            const float* src = P + ((size_t)b * Tsz + t + 1) * NC1 + j * 4;
            float* dst = &xpbuf[(t + 1) & 1][j * 4];
            __builtin_amdgcn_global_load_async_to_lds_b128(
                (as1_v4i*)src, (as3_v4i*)dst, 0, 0);
        }
#endif
        if (tid < Dsz) {
#ifdef ASYNC_LDS
            const float* xp = xpbuf[t & 1];
#else
            size_t prow = ((size_t)b * Tsz + t) * NC1;
            const float* xp = P + prow;
            if (t + 8 < Tsz) __builtin_prefetch(xp + 8 * NC1 + tid, 0, 3);
#endif
            float xr = xp[tid], xz = xp[Dsz + tid], xn = xp[2 * Dsz + tid];
            float hr = gh[tid], hz = gh[Dsz + tid], hn = gh[2 * Dsz + tid];
            float r = 1.0f / (1.0f + expf(-(xr + hr)));
            float z = 1.0f / (1.0f + expf(-(xz + hz)));
            float n = tanhf(xn + r * hn);
            float hNew = (1.0f - z) * n + z * h32[tid];
            h32[tid] = hNew;
            h16[tid] = (_Float16)hNew;
            size_t row = (size_t)b * Tsz + t;
            comb[row * 256 + tid] = (_Float16)hNew;
        }
        __syncthreads();  // barrier 2: h updated before next step's A loads
    }
    if (tid < Dsz) hidden[b * Dsz + tid] = h32[tid];
}

// ---------------------------------------------------------------------------
// Local attention (window 32): one wave per (b,t).
// ---------------------------------------------------------------------------
__global__ __launch_bounds__(256) void k_attn(
    const float* __restrict__ P, _Float16* __restrict__ comb)
{
    int wave = threadIdx.x >> 5, lane = threadIdx.x & 31;
    int g = blockIdx.x * 8 + wave;               // 0..16383
    int b = g >> 12, t = g & (Tsz - 1);

    const float* Q = P + (size_t)g * NC1 + 384;
    int idx = t - (CYC - 1) + lane;
    bool valid = idx >= 0;
    int idxc = valid ? idx : 0;
    const float* K = P + ((size_t)(b * Tsz + idxc)) * NC1 + 512;

    float s = 0.0f;
#pragma unroll
    for (int d = 0; d < Dsz; d += 4) {
        float4 q = *(const float4*)(Q + d);
        float4 k = *(const float4*)(K + d);
        s += q.x * k.x + q.y * k.y + q.z * k.z + q.w * k.w;
    }
    s = valid ? s * (1.0f / 11.313708498984761f) : -__builtin_inff();

    float m = s;
#pragma unroll
    for (int o = 16; o >= 1; o >>= 1) m = fmaxf(m, __shfl_xor(m, o));
    float p = valid ? expf(s - m) : 0.0f;
    float sum = p;
#pragma unroll
    for (int o = 16; o >= 1; o >>= 1) sum += __shfl_xor(sum, o);
    p /= sum;

    int d0 = lane * 4;
    float ax = 0.f, ay = 0.f, az = 0.f, aw = 0.f;
#pragma unroll
    for (int w = 0; w < CYC; w++) {
        float pw = __shfl(p, w);
        int iw = __shfl(idxc, w);
        float4 v = *(const float4*)(P + ((size_t)(b * Tsz + iw)) * NC1 + 640 + d0);
        ax += pw * v.x; ay += pw * v.y; az += pw * v.z; aw += pw * v.w;
    }
    _Float16* o = comb + (size_t)g * 256 + Dsz + d0;
    o[0] = (_Float16)ax; o[1] = (_Float16)ay;
    o[2] = (_Float16)az; o[3] = (_Float16)aw;
}

// ---------------------------------------------------------------------------
// Output projection + residual + LayerNorm, fused per 16-row tile.
// ---------------------------------------------------------------------------
__global__ __launch_bounds__(256) void k_out(
    const _Float16* __restrict__ comb, const _Float16* __restrict__ wp16,
    const float* __restrict__ x, const float* __restrict__ gamma,
    const float* __restrict__ beta, float* __restrict__ out)
{
    int wave = threadIdx.x >> 5, lane = threadIdx.x & 31;
    int lrow = lane & 15, hi = lane >> 4;
    int mt = blockIdx.x;

    __shared__ float res[16][132];   // padded to avoid bank conflicts
    __shared__ float mu_s[16], rs_s[16];

    v8f acc;
#pragma unroll
    for (int i = 0; i < 8; i++) acc[i] = 0.0f;

    const _Float16* arow = comb + (size_t)(mt * 16 + lrow) * 256 + hi * 16;
    const _Float16* brow = wp16 + (size_t)(wave * 16 + lrow) * 256 + hi * 16;
#pragma unroll
    for (int c = 0; c < 8; c++) {
        v16h a  = *(const v16h*)(arow + c * 32);
        v16h bm = *(const v16h*)(brow + c * 32);
        acc = __builtin_amdgcn_wmma_f32_16x16x32_f16(false, a, false, bm,
                                                     (short)0, acc, false, false);
    }
    int col = wave * 16 + lrow;
#pragma unroll
    for (int r = 0; r < 8; r++) {
        int m = r + hi * 8;
        int row = mt * 16 + m;
        res[m][col] = acc[r] + x[(size_t)row * Dsz + col];
    }
    __syncthreads();

    if (threadIdx.x < 16) {
        float su = 0.f, sq = 0.f;
        for (int j = 0; j < Dsz; j++) {
            float v = res[threadIdx.x][j];
            su += v; sq += v * v;
        }
        float mu = su * (1.0f / Dsz);
        float var = sq * (1.0f / Dsz) - mu * mu;
        mu_s[threadIdx.x] = mu;
        rs_s[threadIdx.x] = rsqrtf(var + 1e-5f);
    }
    __syncthreads();

    int rr = threadIdx.x >> 4;
    int j0 = (threadIdx.x & 15) * 8;
    float mu = mu_s[rr], rstd = rs_s[rr];
    size_t row = (size_t)mt * 16 + rr;
#pragma unroll
    for (int i = 0; i < 8; i++) {
        int j = j0 + i;
        out[row * Dsz + j] = (res[rr][j] - mu) * rstd * gamma[j] + beta[j];
    }
}

// ---------------------------------------------------------------------------
extern "C" void kernel_launch(void* const* d_in, const int* in_sizes, int n_in,
                              void* d_out, int out_size, void* d_ws, size_t ws_size,
                              hipStream_t stream)
{
    const float* x     = (const float*)d_in[0];
    const float* h0    = (const float*)d_in[1];
    const float* w_ih  = (const float*)d_in[2];
    const float* w_hh  = (const float*)d_in[3];
    const float* b_ih  = (const float*)d_in[4];
    const float* b_hh  = (const float*)d_in[5];
    const float* wq    = (const float*)d_in[6];
    const float* wk    = (const float*)d_in[7];
    const float* wv    = (const float*)d_in[8];
    const float* wp    = (const float*)d_in[9];
    const float* gamma = (const float*)d_in[10];
    const float* beta  = (const float*)d_in[11];

    float* out    = (float*)d_out;
    float* hidden = out + (size_t)BT * Dsz;

    // workspace carve (all offsets multiples of 256 B)
    uint8_t* w = (uint8_t*)d_ws;
    _Float16* xf16   = (_Float16*)w;  w += (size_t)BT * Dsz * 2;      // 4 MB
    _Float16* wcat   = (_Float16*)w;  w += (size_t)NC1 * Dsz * 2;     // 192 KB
    float*    biascat= (float*)w;     w += (size_t)NC1 * 4;           // 3 KB
    _Float16* wp16   = (_Float16*)w;  w += (size_t)Dsz * 256 * 2;     // 64 KB
    float*    P      = (float*)w;     w += (size_t)BT * NC1 * 4;      // 48 MB
    _Float16* comb   = (_Float16*)w;  w += (size_t)BT * 256 * 2;      // 8 MB

    k_prep<<<(BT * Dsz + 255) / 256, 256, 0, stream>>>(
        x, w_ih, wq, wk, wv, b_ih, wp, xf16, wcat, biascat, wp16);

    k_gemm1<<<(1024 * 48) / 8, 256, 0, stream>>>(xf16, wcat, biascat, P);

    k_gru<<<Bsz, 768, 0, stream>>>(P, w_hh, b_hh, h0, comb, hidden);

    k_attn<<<BT / 8, 256, 0, stream>>>(P, comb);

    k_out<<<BT / 16, 256, 0, stream>>>(comb, wp16, x, gamma, beta, out);
}
